// RoPEMultiHeadAttention_31404800868521
// MI455X (gfx1250) — matrix operand order repaired and checked by reading, not compiled, not run on hardware
//
#include <hip/hip_runtime.h>
#include <hip/hip_bf16.h>

// ---------------------------------------------------------------------------
// RoPE Multi-Head Attention for MI455X (gfx1250, wave32, WMMA bf16).
// B=2, S=2048, D_MODEL=1024, H=16, Dh=64.
// Pipeline: cvt(f32->bf16) -> QKV GEMM (WMMA, async double-buffered LDS) ->
//           RoPE -> flash attention (WMMA, S^T=K*Q^T trick) -> out GEMM.
// ---------------------------------------------------------------------------

typedef __bf16 bf16;
typedef __attribute__((ext_vector_type(16))) __bf16 v16bf;
typedef __attribute__((ext_vector_type(8)))  __bf16 v8bf;
typedef __attribute__((ext_vector_type(4)))  __bf16 v4bf;
typedef __attribute__((ext_vector_type(8)))  float  v8f;
typedef __attribute__((ext_vector_type(4)))  int    v4i;

#define DEV_INLINE __device__ __forceinline__

DEV_INLINE v8f wmma_bf16(v16bf a, v16bf b, v8f c) {
  // D = A(16x32) * B(32x16) + C, f32 accumulate
  return __builtin_amdgcn_wmma_f32_16x16x32_bf16(
      /*neg_a=*/false, a, /*neg_b=*/false, b,
      /*c_mod=*/(short)0, c, /*reuse_a=*/false, /*reuse_b=*/false);
}

// A-operand fragment (16 rows x 32 cols of 16-bit) from a row-major source.
// Lane L (row = L%16) holds K-chunks {koff..koff+7} and {koff+16..koff+23},
// koff = 8 for high half-lanes (per CDNA5 ISA 16-bit A layout).
DEV_INLINE v16bf load_frag_a(const bf16* base, int stride) {
  const int lane = (int)(threadIdx.x & 31u);
  const bf16* p = base + (size_t)(lane & 15) * stride + ((lane & 16) ? 8 : 0);
  v8bf lo = *(const v8bf*)(p);
  v8bf hi = *(const v8bf*)(p + 16);
  return __builtin_shufflevector(lo, hi, 0,1,2,3,4,5,6,7,8,9,10,11,12,13,14,15);
}

// B-operand fragment (32 x 16 of 16-bit) from source stored [n][k] row-major.
// Lane L (col n = L%16) holds K-chunks {koff..koff+7} and {koff+16..koff+23},
// koff = 8 for high half-lanes (same K order as the A fragment).
DEV_INLINE v16bf load_frag_b(const bf16* base, int stride) {
  const int lane = (int)(threadIdx.x & 31u);
  const bf16* p = base + (size_t)(lane & 15) * stride + ((lane & 16) ? 8 : 0);
  v8bf lo = *(const v8bf*)(p);
  v8bf hi = *(const v8bf*)(p + 16);
  return __builtin_shufflevector(lo, hi, 0,1,2,3,4,5,6,7,8,9,10,11,12,13,14,15);
}

// 16-byte global -> LDS copy. Prefer CDNA5 async-to-LDS (ASYNCcnt-tracked);
// fall back to a synchronous register round-trip if the builtin is absent.
// Builtin signature (from hipcc diagnostic): (int4* global, int4* lds, Ii, Ii).
DEV_INLINE void copy16_g2l(const bf16* g, bf16* l) {
#if __has_builtin(__builtin_amdgcn_global_load_async_to_lds_b128)
  __builtin_amdgcn_global_load_async_to_lds_b128((v4i*)g, (v4i*)l, 0, 0);
#else
  *(v8bf*)l = *(const v8bf*)g;
#endif
}

DEV_INLINE void async_wait0() {
#if __has_builtin(__builtin_amdgcn_s_wait_asynccnt)
  __builtin_amdgcn_s_wait_asynccnt(0);
#endif
}

// ---------------------------------------------------------------------------
// fp32 -> bf16 conversion (vectorized x4)
// ---------------------------------------------------------------------------
__global__ __launch_bounds__(256) void cvt_f32_to_bf16(
    const float* __restrict__ in, bf16* __restrict__ out, int n4) {
  int i = (int)(blockIdx.x * 256 + threadIdx.x);
  if (i >= n4) return;
  float4 f = ((const float4*)in)[i];
  v4bf o;
  o[0] = (bf16)f.x; o[1] = (bf16)f.y; o[2] = (bf16)f.z; o[3] = (bf16)f.w;
  ((v4bf*)out)[i] = o;
}

// ---------------------------------------------------------------------------
// Tiled WMMA GEMM: C[M x N] = A[M x K] * W[N x K]^T  (both bf16, f32 acc)
// Block: 256 threads (8 waves), tile 128x128, K-step 32, double-buffered LDS
// filled by GLOBAL_LOAD_ASYNC_TO_LDS_B128. Wave w owns a 32x64 sub-tile:
// rows (w>>1)*32 .. +31, cols (w&1)*64 .. +63  => 8 accumulators, 8 WMMA/step.
// mode 0: write fp32 C row-major (output projection -> d_out)
// mode 1: QKV scatter epilogue: e -> (which,h,dd); q/k head-major, v transposed
// ---------------------------------------------------------------------------
__global__ __launch_bounds__(256) void gemm_wmma_kernel(
    const bf16* __restrict__ A, const bf16* __restrict__ W,
    int M, int N, int K,
    float* __restrict__ Cout,
    bf16* __restrict__ qo, bf16* __restrict__ ko, bf16* __restrict__ vo,
    int mode) {
  __shared__ __align__(128) bf16 As[2][128 * 32];
  __shared__ __align__(128) bf16 Ws[2][128 * 32];
  const int wave = (int)(threadIdx.x >> 5);
  const int lane = (int)(threadIdx.x & 31u);
  const int rgrp = wave >> 1;   // 0..3 -> 32-row group
  const int cgrp = wave & 1;    // 0..1 -> 64-col group
  const int mbase = (int)blockIdx.y * 128;
  const int nbase = (int)blockIdx.x * 128;

  v8f acc[2][4];
#pragma unroll
  for (int a = 0; a < 2; ++a)
#pragma unroll
    for (int nb = 0; nb < 4; ++nb)
#pragma unroll
      for (int j = 0; j < 8; ++j) acc[a][nb][j] = 0.0f;

  // stage K-tile k0 into buffer buf (each thread copies 2x16B per matrix)
  auto stage = [&](int k0, int buf) {
#pragma unroll
    for (int i = 0; i < 2; ++i) {
      int c = (int)threadIdx.x + i * 256;  // 512 chunks of 16B per 128x32 tile
      int row = c >> 2;
      int ch = (c & 3) * 8;
      copy16_g2l(A + (size_t)(mbase + row) * K + k0 + ch,
                 &As[buf][row * 32 + ch]);
      copy16_g2l(W + (size_t)(nbase + row) * K + k0 + ch,
                 &Ws[buf][row * 32 + ch]);
    }
  };

  stage(0, 0);
  int cur = 0;
  for (int k0 = 0; k0 < K; k0 += 32) {
    async_wait0();
    __syncthreads();
    if (k0 + 32 < K) {
      stage(k0 + 32, cur ^ 1);
      // speculative prefetch of the tile after next (global_prefetch_b8)
      if (k0 + 64 < K)
        __builtin_prefetch(
            A + (size_t)(mbase + ((int)threadIdx.x >> 1)) * K + (k0 + 64), 0, 1);
    }
    v16bf af0 = load_frag_a(&As[cur][(rgrp * 32) * 32], 32);
    v16bf af1 = load_frag_a(&As[cur][(rgrp * 32 + 16) * 32], 32);
#pragma unroll
    for (int nb = 0; nb < 4; ++nb) {
      v16bf wf = load_frag_b(&Ws[cur][(cgrp * 64 + nb * 16) * 32], 32);
      acc[0][nb] = wmma_bf16(af0, wf, acc[0][nb]);
      acc[1][nb] = wmma_bf16(af1, wf, acc[1][nb]);
    }
    cur ^= 1;
  }

  const int rhalf = (lane & 16) ? 8 : 0;
  const int nloc = lane & 15;
  if (mode == 0) {
#pragma unroll
    for (int ar = 0; ar < 2; ++ar)
#pragma unroll
      for (int nb = 0; nb < 4; ++nb)
#pragma unroll
        for (int v = 0; v < 8; ++v) {
          int row = mbase + rgrp * 32 + ar * 16 + v + rhalf;
          int col = nbase + cgrp * 64 + nb * 16 + nloc;
          Cout[(size_t)row * N + col] = acc[ar][nb][v];
        }
  } else {
#pragma unroll
    for (int ar = 0; ar < 2; ++ar)
#pragma unroll
      for (int nb = 0; nb < 4; ++nb)
#pragma unroll
        for (int v = 0; v < 8; ++v) {
          int row = mbase + rgrp * 32 + ar * 16 + v + rhalf;  // token [0, B*S)
          int e = nbase + cgrp * 64 + nb * 16 + nloc;         // [0, 3*D_MODEL)
          int which = e >> 10;
          int rem = e & 1023;
          int hh = rem >> 6;
          int dd = rem & 63;
          int bb = row >> 11;
          int ss = row & 2047;
          bf16 val = (bf16)acc[ar][nb][v];
          size_t bhh = (size_t)(bb * 16 + hh);
          if (which == 0)
            qo[(bhh * 2048 + ss) * 64 + dd] = val;  // q: [bh][s][d]
          else if (which == 1)
            ko[(bhh * 2048 + ss) * 64 + dd] = val;  // k: [bh][s][d]
          else
            vo[(bhh * 64 + dd) * 2048 + ss] = val;  // v^T: [bh][d][s]
        }
  }
}

// ---------------------------------------------------------------------------
// RoPE in-place on q,k ([bh][s][64]); pair (j, j+32), angle = s*10000^(-j/32)
// ---------------------------------------------------------------------------
__global__ __launch_bounds__(256) void rope_kernel(bf16* __restrict__ q,
                                                   bf16* __restrict__ k) {
  int idx = (int)(blockIdx.x * 256 + threadIdx.x);  // B*H*S*32 threads
  int j = idx & 31;
  int s = (idx >> 5) & 2047;
  int bh = idx >> 16;
  // ln(10000)/32 = 0.28782313662...
  float ang = (float)s * __expf(-0.28782313662425572f * (float)j);
  float cs = __cosf(ang), sn = __sinf(ang);
  size_t base = ((size_t)bh * 2048 + s) * 64 + j;
  float q0 = (float)q[base], q1 = (float)q[base + 32];
  q[base] = (bf16)(q0 * cs - q1 * sn);
  q[base + 32] = (bf16)(q1 * cs + q0 * sn);
  float k0 = (float)k[base], k1 = (float)k[base + 32];
  k[base] = (bf16)(k0 * cs - k1 * sn);
  k[base + 32] = (bf16)(k1 * cs + k0 * sn);
}

// ---------------------------------------------------------------------------
// Causal flash attention. Grid: (S/64, B*H). Block: 128 threads (4 waves).
// Each wave owns 16 query rows. S^T = K*Q^T so per-query softmax stats are
// per-lane; S^T accumulator layout == A-operand layout of the O = P*V WMMA.
// ---------------------------------------------------------------------------
__global__ __launch_bounds__(128) void flash_attn_kernel(
    const bf16* __restrict__ Q, const bf16* __restrict__ Km,
    const bf16* __restrict__ Vt, bf16* __restrict__ Out) {
  const int S = 2048;
  const int bh = (int)blockIdx.y;
  const int b = bh >> 4;
  const int h = bh & 15;
  const int wave = (int)(threadIdx.x >> 5);
  const int lane = (int)(threadIdx.x & 31u);
  const int hi = (lane & 16) ? 1 : 0;
  const int qbase = (int)blockIdx.x * 64 + wave * 16;

  const bf16* qb = Q + (size_t)bh * S * 64;
  const bf16* kb = Km + (size_t)bh * S * 64;
  const bf16* vtb = Vt + (size_t)bh * 64 * S;

  // Q^T B-operand fragments (d 0..31, d 32..63) — loaded once
  v16bf qf0 = load_frag_b(qb + (size_t)qbase * 64, 64);
  v16bf qf1 = load_frag_b(qb + (size_t)qbase * 64 + 32, 64);

  v8f o[4];
#pragma unroll
  for (int t = 0; t < 4; ++t)
#pragma unroll
    for (int j = 0; j < 8; ++j) o[t][j] = 0.0f;

  float m = -3.0e38f, l = 0.0f;
  const int q_lane = qbase + (lane & 15);
  const float scale = 0.125f;          // 1/sqrt(64)
  const int nblk = (qbase + 47) >> 5;  // causal: k <= qbase+15

  for (int blk = 0; blk < nblk; ++blk) {
    const int kb0 = blk * 32;
    v8f st0, st1;
#pragma unroll
    for (int j = 0; j < 8; ++j) { st0[j] = 0.0f; st1[j] = 0.0f; }

    v16bf ka;
    ka = load_frag_a(kb + (size_t)kb0 * 64, 64);
    st0 = wmma_bf16(ka, qf0, st0);
    ka = load_frag_a(kb + (size_t)kb0 * 64 + 32, 64);
    st0 = wmma_bf16(ka, qf1, st0);
    ka = load_frag_a(kb + (size_t)(kb0 + 16) * 64, 64);
    st1 = wmma_bf16(ka, qf0, st1);
    ka = load_frag_a(kb + (size_t)(kb0 + 16) * 64 + 32, 64);
    st1 = wmma_bf16(ka, qf1, st1);

    // scale + causal mask + per-query running max (lane holds query q_lane)
    float s0[8], s1[8];
    float mloc = -3.0e38f;
#pragma unroll
    for (int v = 0; v < 8; ++v) {
      int k0i = kb0 + v + hi * 8;
      int k1i = k0i + 16;
      float x0 = (k0i <= q_lane) ? st0[v] * scale : -3.0e38f;
      float x1 = (k1i <= q_lane) ? st1[v] * scale : -3.0e38f;
      s0[v] = x0; s1[v] = x1;
      mloc = fmaxf(mloc, fmaxf(x0, x1));
    }
    mloc = fmaxf(mloc, __shfl_xor(mloc, 16, 32));
    const float mnew = fmaxf(m, mloc);
    const float rescale = __expf(m - mnew);

    float psum = 0.0f;
    v16bf pf;  // A-operand fragment of P — matches S^T accumulator layout
#pragma unroll
    for (int v = 0; v < 8; ++v) {
      float p0 = (s0[v] > -1.0e37f) ? __expf(s0[v] - mnew) : 0.0f;
      float p1 = (s1[v] > -1.0e37f) ? __expf(s1[v] - mnew) : 0.0f;
      psum += p0 + p1;
      pf[v] = (bf16)p0;
      pf[v + 8] = (bf16)p1;
    }
    psum += __shfl_xor(psum, 16, 32);
    l = l * rescale + psum;
    m = mnew;

    // rescale O accumulators: O tile rows are queries (v + 8*hi)
#pragma unroll
    for (int v = 0; v < 8; ++v) {
      float rv = __shfl(rescale, v + hi * 8, 32);
      o[0][v] *= rv; o[1][v] *= rv; o[2][v] *= rv; o[3][v] *= rv;
    }
    // O += P(16q x 32k) * V(32k x 16d) for 4 d-blocks (V^T rows contiguous)
#pragma unroll
    for (int t = 0; t < 4; ++t) {
      v16bf vf = load_frag_b(vtb + (size_t)(t * 16) * S + kb0, S);
      o[t] = wmma_bf16(pf, vf, o[t]);
    }
  }

  const float invl = 1.0f / l;
#pragma unroll
  for (int v = 0; v < 8; ++v) {
    float iv = __shfl(invl, v + hi * 8, 32);
    int s = qbase + v + hi * 8;
    size_t base = ((size_t)(b * S + s)) * 1024 + (size_t)h * 64 + (lane & 15);
#pragma unroll
    for (int t = 0; t < 4; ++t) Out[base + t * 16] = (bf16)(o[t][v] * iv);
  }
}

// ---------------------------------------------------------------------------
// Host launch. Workspace layout (bytes, ~48 MB total):
//   0        x_bf16      (4096*1024*2 = 8 MB)
//   8  MB    wqkv_bf16   (3072*1024*2 = 6 MB)
//   14 MB    wout_bf16   (1024*1024*2 = 2 MB)
//   16 MB    q  [bh][s][d]   8 MB
//   24 MB    k  [bh][s][d]   8 MB
//   32 MB    v^T[bh][d][s]   8 MB
//   40 MB    attn out [b][s][h*d] bf16  8 MB
// ---------------------------------------------------------------------------
extern "C" void kernel_launch(void* const* d_in, const int* in_sizes, int n_in,
                              void* d_out, int out_size, void* d_ws,
                              size_t ws_size, hipStream_t stream) {
  (void)in_sizes; (void)n_in; (void)out_size; (void)ws_size;
  const float* x = (const float*)d_in[0];
  const float* w_qkv = (const float*)d_in[1];
  const float* w_out = (const float*)d_in[2];
  float* out = (float*)d_out;
  char* ws = (char*)d_ws;

  bf16* xb    = (bf16*)(ws + 0);
  bf16* wqkvb = (bf16*)(ws + 8388608u);
  bf16* woutb = (bf16*)(ws + 14680064u);
  bf16* qbuf  = (bf16*)(ws + 16777216u);
  bf16* kbuf  = (bf16*)(ws + 25165824u);
  bf16* vtbuf = (bf16*)(ws + 33554432u);
  bf16* aobuf = (bf16*)(ws + 41943040u);

  cvt_f32_to_bf16<<<dim3(4096), 256, 0, stream>>>(x, xb, 4096 * 1024 / 4);
  cvt_f32_to_bf16<<<dim3(3072), 256, 0, stream>>>(w_qkv, wqkvb, 3072 * 1024 / 4);
  cvt_f32_to_bf16<<<dim3(1024), 256, 0, stream>>>(w_out, woutb, 1024 * 1024 / 4);

  // QKV projection: M=4096 tokens, N=3072, K=1024, scatter epilogue
  gemm_wmma_kernel<<<dim3(24, 32), 256, 0, stream>>>(
      xb, wqkvb, 4096, 3072, 1024, nullptr, qbuf, kbuf, vtbuf, 1);

  rope_kernel<<<dim3(8192), 256, 0, stream>>>(qbuf, kbuf);

  flash_attn_kernel<<<dim3(32, 32), 128, 0, stream>>>(qbuf, kbuf, vtbuf, aobuf);

  // Output projection: M=4096, N=1024, K=1024, fp32 out
  gemm_wmma_kernel<<<dim3(8, 32), 256, 0, stream>>>(
      aobuf, woutb, 4096, 1024, 1024, out, nullptr, nullptr, nullptr, 0);
}
